// HybridBlock_33363305955333
// MI455X (gfx1250) — matrix operand order repaired and checked by reading, not compiled, not run on hardware
//
#include <hip/hip_runtime.h>

// ---------------- problem constants ----------------
#define BB 4
#define TT 2048
#define DD 1024
#define HH 8192
#define MM (BB * TT)          // 8192 rows
#define THRESH 0.05f

typedef __attribute__((ext_vector_type(16))) __bf16 v16bf;
typedef __attribute__((ext_vector_type(8)))  __bf16 v8bf;
typedef __attribute__((ext_vector_type(8)))  float  v8f;
typedef __attribute__((ext_vector_type(4)))  float  f32x4;
typedef __attribute__((ext_vector_type(4)))  unsigned int v4u;
typedef __attribute__((ext_vector_type(8)))  int v8i;
typedef __attribute__((ext_vector_type(4)))  int v4i;

__device__ __forceinline__ v8f wmma_bf16(v16bf a, v16bf b, v8f c) {
    return __builtin_amdgcn_wmma_f32_16x16x32_bf16(false, a, false, b,
                                                   (short)0, c, false, false);
}

__device__ __forceinline__ float sigmoidf_fast(float x) {
    return 1.f / (1.f + __expf(-x));
}

// ---------------- Tensor Data Mover: 2D bf16 tile -> LDS ----------------
// Loads a (rows x 32) bf16 tile whose rows are K-strided in global memory into
// LDS with a 4-DWORD pad every 16 DWORDs (row pitch 64B -> 80B = 40 elements),
// matching the bank-conflict-free layout the WMMA fragment loads expect.
// D# packing per CDNA5 ISA.
__device__ __forceinline__ void tdm_load_tile(unsigned lds_addr,
                                              const __bf16* gptr,
                                              int rows, int stride_elems) {
    const unsigned long long ga = (unsigned long long)(uintptr_t)gptr;
    v4u g0;
    g0[0] = 1u;                                   // count=1, user descriptor
    g0[1] = lds_addr;                             // LDS byte address
    g0[2] = (unsigned)(ga & 0xffffffffu);         // global_addr[31:0]
    g0[3] = (unsigned)((ga >> 32) & 0x01ffffffu)  // global_addr[56:32]
            | (2u << 30);                         // type = 2 ("image")
    const unsigned w0 = (1u << 16)                // data_size = 1 -> 2 bytes
                      | (1u << 20)                // pad_enable
                      | (3u << 22)                // pad_interval: 16 DWORDs
                      | (3u << 25);               // pad_amount:   4 DWORDs
    const unsigned td0 = 32u;                     // tensor_dim0 (elements)
    const unsigned td1 = (unsigned)rows;          // tensor_dim1
    const unsigned long long s0 = (unsigned long long)stride_elems;
    v8i g1;
    g1[0] = (int)w0;
    g1[1] = (int)((td0 & 0xffffu) << 16);                         // dim0 lo16
    g1[2] = (int)(((td0 >> 16) & 0xffffu) | ((td1 & 0xffffu) << 16));
    g1[3] = (int)(((td1 >> 16) & 0xffffu) | (32u << 16));         // tile_dim0=32
    g1[4] = (int)((unsigned)rows & 0xffffu);                      // tile_dim1
    g1[5] = (int)(unsigned)(s0 & 0xffffffffu);                    // stride0 lo
    g1[6] = (int)(unsigned)((s0 >> 32) & 0xffffu);                // stride0 hi
    g1[7] = 0;
    const v4i z4 = {0, 0, 0, 0};
#if __clang_major__ >= 23
    const v8i z8 = {0, 0, 0, 0, 0, 0, 0, 0};
    __builtin_amdgcn_tensor_load_to_lds(g0, g1, z4, z4, z8, 0);
#else
    __builtin_amdgcn_tensor_load_to_lds(g0, g1, z4, z4, 0);
#endif
}

__device__ __forceinline__ unsigned lds_off(const void* p) {
    return (unsigned)(uintptr_t)p;   // low 32 bits of generic ptr = LDS offset
}

// ---------------- elementwise converters ----------------
__global__ void f32_to_bf16_kernel(const float* __restrict__ in,
                                   __bf16* __restrict__ out, int n) {
    for (int i = blockIdx.x * blockDim.x + threadIdx.x; i < n;
         i += gridDim.x * blockDim.x)
        out[i] = (__bf16)in[i];
}

__global__ void syn_mask_bf16_kernel(const float* __restrict__ w,
                                     const unsigned char* __restrict__ mask,
                                     __bf16* __restrict__ out, int n) {
    for (int i = blockIdx.x * blockDim.x + threadIdx.x; i < n;
         i += gridDim.x * blockDim.x)
        out[i] = (__bf16)(mask[i] ? w[i] : 0.f);
}

// ---------------- rmsnorm -> bf16 ----------------
__global__ __launch_bounds__(256)
void rmsnorm_bf16_kernel(const float* __restrict__ x,
                         const float* __restrict__ w,
                         __bf16* __restrict__ out) {
    const int row = blockIdx.x;
    const int tid = threadIdx.x;
    const float* xr = x + (size_t)row * DD;
    f32x4 v = ((const f32x4*)xr)[tid];            // 256 thr * 4 = 1024
    float s = v[0]*v[0] + v[1]*v[1] + v[2]*v[2] + v[3]*v[3];
#pragma unroll
    for (int o = 16; o > 0; o >>= 1) s += __shfl_xor(s, o, 32);
    __shared__ float red[8];
    const int lane = tid & 31, wave = tid >> 5;
    if (lane == 0) red[wave] = s;
    __syncthreads();
    if (tid == 0) {
        float tot = 0.f;
#pragma unroll
        for (int i = 0; i < 8; ++i) tot += red[i];
        red[0] = rsqrtf(tot * (1.f / (float)DD) + 1e-6f);
    }
    __syncthreads();
    const float scale = red[0];
    const float* wr = w + tid * 4;
    __bf16* orow = out + (size_t)row * DD + tid * 4;
#pragma unroll
    for (int i = 0; i < 4; ++i) orow[i] = (__bf16)(v[i] * scale * wr[i]);
}

// ---------------- init (state + global barrier) ----------------
__global__ void init_kernel(__bf16* __restrict__ h, int* __restrict__ bar) {
    int i = blockIdx.x * blockDim.x + threadIdx.x;
    if (i < 16 * DD) h[i] = (__bf16)0.f;
    if (i == 0) *bar = 0;
}

// ---------------- tiled bf16 WMMA GEMM, C = A(M,K) x B(N,K)^T ----------------
// TDM feeds double-buffered LDS tiles; wave 0 issues tensor_load_to_lds for
// tile kt+1 while all 8 waves run WMMA on tile kt. One barrier per K-step.
// Waves form a 4(M) x 2(N) grid: each wave computes 32 rows and BN/2 cols,
// giving 2-way in-wave reuse of every B fragment.
// EPI 0: outF = acc                                (P projection)
// EPI 1: outF = resid + acc0 * sigmoid(acc1+bias)  (sparse-syn * gate, dual B)
// EPI 2: outH = bf16(silu(acc0) * acc1)            (SwiGLU gate*up, dual B)
// EPI 3: outF = resid + acc0                       (down-proj + residual)
template <int EPI>
__global__ __launch_bounds__(256)
void gemm_bf16_kernel(const __bf16* __restrict__ A,
                      const __bf16* __restrict__ B0,
                      const __bf16* __restrict__ B1,
                      const float* __restrict__ resid,
                      const float* __restrict__ bias,
                      float* __restrict__ outF,
                      __bf16* __restrict__ outH,
                      int K, int N) {
    constexpr bool DUAL = (EPI == 1 || EPI == 2);
    constexpr int BM = 128;
    constexpr int BN = DUAL ? 64 : 128;
    constexpr int BK = 32;
    constexpr int NTW = DUAL ? 2 : 4;  // 16-col tiles per wave
    constexpr int NW  = BN / 2;        // cols per wave
    constexpr int LP = 40;             // padded LDS row pitch (elements)

    __shared__ __attribute__((aligned(16))) __bf16 As [2][BM * LP];
    __shared__ __attribute__((aligned(16))) __bf16 Bs0[2][BN * LP];
    __shared__ __attribute__((aligned(16))) __bf16 Bs1[DUAL ? 2 : 1][DUAL ? BN * LP : 8];

    const int tid  = threadIdx.x;
    const int wave = tid >> 5;
    const int lane = tid & 31;
    const int l16  = lane & 15;
    const int lh   = lane >> 4;
    const int mw   = wave >> 1;        // 0..3  (M strip of 32 rows)
    const int nw   = wave & 1;         // 0..1  (N half)

    v8f acc0[2][NTW];
    v8f acc1[DUAL ? 2 : 1][DUAL ? NTW : 1];
    const v8f vz = {0.f, 0.f, 0.f, 0.f, 0.f, 0.f, 0.f, 0.f};
#pragma unroll
    for (int mt = 0; mt < 2; ++mt)
#pragma unroll
        for (int nt = 0; nt < NTW; ++nt) acc0[mt][nt] = vz;
    if (DUAL) {
#pragma unroll
        for (int mt = 0; mt < 2; ++mt)
#pragma unroll
            for (int nt = 0; nt < NTW; ++nt) acc1[mt][nt] = vz;
    }

    const int rowBase = blockIdx.y * BM;
    const int colBase = blockIdx.x * BN;
    const int nk = K / BK;

    // prologue: async-load tile 0 into buffer 0
    if (wave == 0) {
        tdm_load_tile(lds_off(&As[0][0]),  A  + (size_t)rowBase * K, BM, K);
        tdm_load_tile(lds_off(&Bs0[0][0]), B0 + (size_t)colBase * K, BN, K);
        if (DUAL)
            tdm_load_tile(lds_off(&Bs1[0][0]), B1 + (size_t)colBase * K, BN, K);
    }

    for (int kt = 0; kt < nk; ++kt) {
        const int cur = kt & 1;
        if (wave == 0) __builtin_amdgcn_s_wait_tensorcnt((short)0);
        __syncthreads();   // tile kt ready; prev compute on other buffer done

        // prefetch next tile into the other buffer while computing this one
        if (wave == 0 && (kt + 1) < nk) {
            const int nxt = cur ^ 1;
            const size_t ko = (size_t)(kt + 1) * BK;
            tdm_load_tile(lds_off(&As[nxt][0]),
                          A + (size_t)rowBase * K + ko, BM, K);
            tdm_load_tile(lds_off(&Bs0[nxt][0]),
                          B0 + (size_t)colBase * K + ko, BN, K);
            if (DUAL)
                tdm_load_tile(lds_off(&Bs1[nxt][0]),
                              B1 + (size_t)colBase * K + ko, BN, K);
        }

        // ---- compute: wave owns rows [mw*32, mw*32+32), cols [nw*NW, +NW) ----
        union { v16bf v; v8bf h[2]; } av[2];
#pragma unroll
        for (int mt = 0; mt < 2; ++mt) {
            const int ar = mw * 32 + mt * 16 + l16;
            av[mt].h[0] = *(const v8bf*)&As[cur][ar * LP + lh * 16];
            av[mt].h[1] = *(const v8bf*)&As[cur][ar * LP + lh * 16 + 8];
        }
#pragma unroll
        for (int nt = 0; nt < NTW; ++nt) {
            const int bn = nw * NW + nt * 16 + l16;
            union { v16bf v; v8bf h[2]; } bv;
            bv.h[0] = *(const v8bf*)&Bs0[cur][bn * LP + lh * 16];
            bv.h[1] = *(const v8bf*)&Bs0[cur][bn * LP + lh * 16 + 8];
#pragma unroll
            for (int mt = 0; mt < 2; ++mt)
                acc0[mt][nt] = wmma_bf16(av[mt].v, bv.v, acc0[mt][nt]);
            if (DUAL) {
                union { v16bf v; v8bf h[2]; } cv;
                cv.h[0] = *(const v8bf*)&Bs1[cur][bn * LP + lh * 16];
                cv.h[1] = *(const v8bf*)&Bs1[cur][bn * LP + lh * 16 + 8];
#pragma unroll
                for (int mt = 0; mt < 2; ++mt)
                    acc1[mt][nt] = wmma_bf16(av[mt].v, cv.v, acc1[mt][nt]);
            }
        }
    }

    // ---- epilogue ----
#pragma unroll
    for (int mt = 0; mt < 2; ++mt) {
#pragma unroll
        for (int nt = 0; nt < NTW; ++nt) {
            const int col = colBase + nw * NW + nt * 16 + l16;
#pragma unroll
            for (int r = 0; r < 8; ++r) {
                const int row = rowBase + mw * 32 + mt * 16 + r + lh * 8;
                const size_t idx = (size_t)row * N + col;
                if (EPI == 0) {
                    outF[idx] = acc0[mt][nt][r];
                } else if (EPI == 1) {
                    const float g = acc1[mt][nt][r] + bias[col];
                    outF[idx] = resid[idx] + acc0[mt][nt][r] * sigmoidf_fast(g);
                } else if (EPI == 2) {
                    const float g = acc0[mt][nt][r];
                    outH[idx] = (__bf16)(g * sigmoidf_fast(g) * acc1[mt][nt][r]);
                } else {
                    outF[idx] = resid[idx] + acc0[mt][nt][r];
                }
            }
        }
    }
}

// ---------------- persistent liquid + PLIF recurrence ----------------
// 16 workgroups x 256 threads, each owns 64 output columns. Wrec WMMA
// fragments are hoisted into registers (loop invariant); per step only the
// 4x1024 bf16 h vector streams from L2. K split over 8 waves, partials merged
// with ds_add_f32, then tanh/liquid + PLIF update, device-wide barrier.
#define NWG_REC 16
#define RCOLS   64

__global__ __launch_bounds__(256)
void recurrence_kernel(const float* __restrict__ P,      // (B*T, D)
                       const __bf16* __restrict__ Wrec,  // (D, D)
                       const float* __restrict__ liq_b,
                       const float* __restrict__ liq_tau,
                       const float* __restrict__ plif_w,
                       __bf16* __restrict__ h_bf,        // 16 x D, rows 4..15 = 0
                       __bf16* __restrict__ s_out,       // (B*T, D)
                       int* __restrict__ bar) {
    __shared__ float accbuf[RCOLS * 16];   // [n][m]
    const int tid  = threadIdx.x;
    const int wave = tid >> 5;
    const int lane = tid & 31;
    const int l16  = lane & 15;
    const int lh   = lane >> 4;
    const int col0 = blockIdx.x * RCOLS;

    for (int i = tid; i < RCOLS * 16; i += 256) accbuf[i] = 0.f;

    // epilogue ownership: thread -> (batch eb, column ed); state in registers
    const int eb = tid >> 6;
    const int ec = tid & 63;
    const int ed = col0 + ec;
    float tauv;
    {
        const float xv = liq_tau[ed];
        tauv = (xv > 20.f) ? xv : log1pf(__expf(xv));
    }
    const float biasv = liq_b[ed];
    const float decay = sigmoidf_fast(plif_w[0]);
    float hst = 0.f, vst = 0.f;

    union FV { v16bf v; v8bf h[2]; };

    // hoist loop-invariant Wrec fragments: 4 k-tiles x 4 n-tiles
    FV bfr[4][4];
#pragma unroll
    for (int kk = 0; kk < 4; ++kk) {
        const int kbase = (wave * 4 + kk) * 32;
#pragma unroll
        for (int nt = 0; nt < 4; ++nt) {
            const int n = col0 + nt * 16 + l16;
            const __bf16* bp = Wrec + (size_t)n * DD + kbase + lh * 16;
            bfr[kk][nt].h[0] = *(const v8bf*)bp;
            bfr[kk][nt].h[1] = *(const v8bf*)(bp + 8);
        }
    }

    const __bf16* Arow = h_bf + (size_t)l16 * DD;   // padded-M A operand rows
    const v8f vz = {0.f, 0.f, 0.f, 0.f, 0.f, 0.f, 0.f, 0.f};

    __syncthreads();

    for (int t = 0; t < TT; ++t) {
        // ---- load all A fragments first (overlap loads with WMMA chain) ----
        FV av[4];
#pragma unroll
        for (int kk = 0; kk < 4; ++kk) {
            const int kbase = (wave * 4 + kk) * 32;
            av[kk].h[0] = *(const v8bf*)(Arow + kbase + lh * 16);
            av[kk].h[1] = *(const v8bf*)(Arow + kbase + lh * 16 + 8);
        }
        // ---- partial GEMM over this wave's K range ----
        v8f acc[4] = {vz, vz, vz, vz};
#pragma unroll
        for (int kk = 0; kk < 4; ++kk) {
#pragma unroll
            for (int nt = 0; nt < 4; ++nt)
                acc[nt] = wmma_bf16(av[kk].v, bfr[kk][nt].v, acc[nt]);
        }
        // ---- cross-wave reduce (only batch rows m<4 matter) ----
#pragma unroll
        for (int nt = 0; nt < 4; ++nt) {
            const int n = nt * 16 + l16;
#pragma unroll
            for (int r = 0; r < 4; ++r) {
                if (lh == 0) atomicAdd(&accbuf[n * 16 + r], acc[nt][r]);
            }
        }
        __syncthreads();

        // ---- liquid + PLIF update for owned (eb, ed) ----
        const float pre = accbuf[ec * 16 + eb] +
                          P[((size_t)eb * TT + t) * DD + ed] + biasv;
        __syncthreads();                      // all reads done before reset
        for (int i = tid; i < RCOLS * 16; i += 256) accbuf[i] = 0.f;

        const float hn = hst + (tanhf(pre) - hst) / tauv;
        vst += decay * (hn - vst);
        const float sp = (vst - THRESH > 0.f) ? 1.f : 0.f;
        vst -= sp * THRESH;
        hst = hn;
        h_bf[(size_t)eb * DD + ed] = (__bf16)hn;
        s_out[((size_t)eb * TT + t) * DD + ed] = (__bf16)sp;

        // ---- device-wide barrier (publish h, then sync all 16 WGs) ----
        __threadfence();
        __syncthreads();
        if (tid == 0) {
            __hip_atomic_fetch_add(bar, 1, __ATOMIC_RELEASE,
                                   __HIP_MEMORY_SCOPE_AGENT);
            const int target = NWG_REC * (t + 1);
            while (__hip_atomic_load(bar, __ATOMIC_ACQUIRE,
                                     __HIP_MEMORY_SCOPE_AGENT) < target)
                __builtin_amdgcn_s_sleep(2);
        }
        __syncthreads();
    }
}

// ---------------- host launcher ----------------
extern "C" void kernel_launch(void* const* d_in, const int* in_sizes, int n_in,
                              void* d_out, int out_size, void* d_ws,
                              size_t ws_size, hipStream_t stream) {
    const float* x        = (const float*)d_in[0];
    const float* ln1_w    = (const float*)d_in[1];
    const float* liq_win  = (const float*)d_in[2];
    const float* liq_wrec = (const float*)d_in[3];
    const float* liq_b    = (const float*)d_in[4];
    const float* liq_tau  = (const float*)d_in[5];
    const float* plif_w   = (const float*)d_in[6];
    const float* syn_w    = (const float*)d_in[7];
    const float* gate_w   = (const float*)d_in[8];
    const float* gate_b   = (const float*)d_in[9];
    const float* ln2_w    = (const float*)d_in[10];
    const float* ffn_wg   = (const float*)d_in[11];
    const float* ffn_wu   = (const float*)d_in[12];
    const float* ffn_wd   = (const float*)d_in[13];
    const unsigned char* syn_mask = (const unsigned char*)d_in[14];

    size_t off = 0;
    auto alloc = [&](size_t bytes) -> void* {
        void* p = (char*)d_ws + off;
        off += (bytes + 255) & ~(size_t)255;
        return p;
    };
    __bf16* Win_bf  = (__bf16*)alloc((size_t)DD * DD * 2);
    __bf16* Wrec_bf = (__bf16*)alloc((size_t)DD * DD * 2);
    __bf16* Syn_bf  = (__bf16*)alloc((size_t)DD * DD * 2);
    __bf16* Gate_bf = (__bf16*)alloc((size_t)DD * DD * 2);
    __bf16* Wg_bf   = (__bf16*)alloc((size_t)HH * DD * 2);
    __bf16* Wu_bf   = (__bf16*)alloc((size_t)HH * DD * 2);
    __bf16* Wd_bf   = (__bf16*)alloc((size_t)DD * HH * 2);
    __bf16* a_bf    = (__bf16*)alloc((size_t)MM * DD * 2);
    float*  Pbuf    = (float*) alloc((size_t)MM * DD * 4);
    __bf16* s_bf    = (__bf16*)alloc((size_t)MM * DD * 2);
    float*  xmid    = (float*) alloc((size_t)MM * DD * 4);
    __bf16* y_bf    = (__bf16*)alloc((size_t)MM * DD * 2);
    __bf16* G_bf    = (__bf16*)alloc((size_t)MM * HH * 2);
    __bf16* h_bf    = (__bf16*)alloc((size_t)16 * DD * 2);
    int*    bar     = (int*)   alloc(256);

    const int DW = DD * DD;   // 1M
    const int HW = HH * DD;   // 8M

    // weight conversions to bf16 (one-time per call; tiny vs GEMMs)
    f32_to_bf16_kernel<<<1024, 256, 0, stream>>>(liq_win, Win_bf, DW);
    f32_to_bf16_kernel<<<1024, 256, 0, stream>>>(liq_wrec, Wrec_bf, DW);
    syn_mask_bf16_kernel<<<1024, 256, 0, stream>>>(syn_w, syn_mask, Syn_bf, DW);
    f32_to_bf16_kernel<<<1024, 256, 0, stream>>>(gate_w, Gate_bf, DW);
    f32_to_bf16_kernel<<<4096, 256, 0, stream>>>(ffn_wg, Wg_bf, HW);
    f32_to_bf16_kernel<<<4096, 256, 0, stream>>>(ffn_wu, Wu_bf, HW);
    f32_to_bf16_kernel<<<4096, 256, 0, stream>>>(ffn_wd, Wd_bf, HW);

    // a = rmsnorm(x, ln1_w) -> bf16
    rmsnorm_bf16_kernel<<<MM, 256, 0, stream>>>(x, ln1_w, a_bf);

    // P = a @ Win^T  (f32 out)
    gemm_bf16_kernel<0><<<dim3(DD / 128, MM / 128), 256, 0, stream>>>(
        a_bf, Win_bf, nullptr, nullptr, nullptr, Pbuf, nullptr, DD, DD);

    // recurrent liquid + PLIF -> spikes
    init_kernel<<<64, 256, 0, stream>>>(h_bf, bar);
    recurrence_kernel<<<NWG_REC, 256, 0, stream>>>(Pbuf, Wrec_bf, liq_b,
                                                   liq_tau, plif_w, h_bf,
                                                   s_bf, bar);

    // xmid = x + (s@syn^T) * sigmoid(s@gate^T + gate_b)
    gemm_bf16_kernel<1><<<dim3(DD / 64, MM / 128), 256, 0, stream>>>(
        s_bf, Syn_bf, Gate_bf, x, gate_b, xmid, nullptr, DD, DD);

    // y = rmsnorm(xmid, ln2_w) -> bf16
    rmsnorm_bf16_kernel<<<MM, 256, 0, stream>>>(xmid, ln2_w, y_bf);

    // G = silu(y@Wg^T) * (y@Wu^T) -> bf16
    gemm_bf16_kernel<2><<<dim3(HH / 64, MM / 128), 256, 0, stream>>>(
        y_bf, Wg_bf, Wu_bf, nullptr, nullptr, nullptr, G_bf, DD, HH);

    // out = xmid + G @ Wd^T
    gemm_bf16_kernel<3><<<dim3(DD / 128, MM / 128), 256, 0, stream>>>(
        G_bf, Wd_bf, nullptr, xmid, nullptr, (float*)d_out, nullptr, HH, DD);

    (void)in_sizes; (void)n_in; (void)out_size; (void)ws_size;
}
